// SlotAttention_83940840833852
// MI455X (gfx1250) — matrix operand order repaired and checked by reading, not compiled, not run on hardware
//
#include <hip/hip_runtime.h>
#include <hip/hip_bf16.h>

typedef __attribute__((ext_vector_type(16))) _Float16 v16h;
typedef __attribute__((ext_vector_type(8)))  float    v8f;
typedef __attribute__((ext_vector_type(4)))  int      v4i;

typedef __attribute__((address_space(1))) v4i g_v4i;   // global int4
typedef __attribute__((address_space(3))) v4i l_v4i;   // LDS int4

#define LN_EPS 0.001f
#define EPS_W  1e-8f

#if defined(__HIP_DEVICE_COMPILE__) && \
    __has_builtin(__builtin_amdgcn_global_load_async_to_lds_b128) && \
    __has_builtin(__builtin_amdgcn_global_store_async_from_lds_b128)
#define HAVE_ASYNC_COPY 1
#else
#define HAVE_ASYNC_COPY 0
#endif

__device__ __forceinline__ void wait_asynccnt0() {
#if defined(__HIP_DEVICE_COMPILE__) && __has_builtin(__builtin_amdgcn_s_wait_asynccnt)
    __builtin_amdgcn_s_wait_asynccnt(0);
#else
    asm volatile("s_wait_asynccnt 0x0" ::: "memory");
#endif
}

// ---------------------------------------------------------------------------
// Kernel 0: init slots (4 x 64).  row 0 = bg, rows 1..3 = fg.
// ---------------------------------------------------------------------------
__global__ __launch_bounds__(256) void init_slots_kernel(
    const float* __restrict__ noise_fg, const float* __restrict__ noise_bg,
    const float* __restrict__ mu_fg, const float* __restrict__ ls_fg,
    const float* __restrict__ mu_bg, const float* __restrict__ ls_bg,
    float* __restrict__ slots)
{
    int tid = threadIdx.x;          // 0..255
    int s = tid >> 6, d = tid & 63;
    float val;
    if (s == 0) val = mu_bg[d] + __expf(ls_bg[d]) * noise_bg[d];
    else        val = mu_fg[d] + __expf(ls_fg[d]) * noise_fg[(s - 1) * 64 + d];
    slots[tid] = val;
}

// ---------------------------------------------------------------------------
// Kernel 1: fused LayerNorm(x) -> k = xn@Wk, v = xn@Wv  (WMMA f16, f32 acc)
// Block: 256 threads (8 waves), 128 rows per block. Outputs stored f16 via
// LDS restage + async coalesced b128 stores.
// ---------------------------------------------------------------------------
__global__ __launch_bounds__(256) void kv_wmma_kernel(
    const float* __restrict__ x,
    const float* __restrict__ ln_g, const float* __restrict__ ln_b,
    const float* __restrict__ Wk, const float* __restrict__ Wv,
    _Float16* __restrict__ k_out, _Float16* __restrict__ v_out)
{
    __shared__ _Float16 xn[128][64];      // normalized input, reused as k stage (16 KB)
    __shared__ _Float16 wkT[64][64];      // Wk transposed [d][c]   (8 KB)
    __shared__ _Float16 wvT[64][64];      // Wv transposed [d][c]   (8 KB)
    __shared__ _Float16 vstage[128][64];  // v output stage (16 KB)

    const int tid  = threadIdx.x;
    const int row0 = blockIdx.x * 128;

    // stage weights (transposed so B fragments are K-contiguous)
    for (int i = tid; i < 64 * 64; i += 256) {
        int c = i >> 6, d = i & 63;
        wkT[d][c] = (_Float16)Wk[i];
        wvT[d][c] = (_Float16)Wv[i];
    }

    // ---- layernorm: 2 threads per row, 32 cols each ----
    {
        int r = tid >> 1;         // 0..127
        int hh = tid & 1;         // which half of the row
        const float* xr = x + (size_t)(row0 + r) * 64 + hh * 32;
        float vals[32];
        float s1 = 0.f, s2 = 0.f;
#pragma unroll
        for (int i = 0; i < 8; ++i) {
            float4 t = reinterpret_cast<const float4*>(xr)[i];
            vals[4*i+0] = t.x; vals[4*i+1] = t.y; vals[4*i+2] = t.z; vals[4*i+3] = t.w;
            s1 += t.x + t.y + t.z + t.w;
            s2 += t.x*t.x + t.y*t.y + t.z*t.z + t.w*t.w;
        }
        s1 += __shfl_xor(s1, 1, 32);
        s2 += __shfl_xor(s2, 1, 32);
        float mean = s1 * (1.f / 64.f);
        float var  = s2 * (1.f / 64.f) - mean * mean;
        float inv  = rsqrtf(var + LN_EPS);
#pragma unroll
        for (int i = 0; i < 32; ++i) {
            int c = hh * 32 + i;
            xn[r][c] = (_Float16)(ln_g[c] * (vals[i] - mean) * inv + ln_b[c]);
        }
    }
    __syncthreads();

    // ---- WMMA: each wave computes a 16x64 tile of k and v ----
    const int wave  = tid >> 5;
    const int lane  = tid & 31;
    const int m16   = lane & 15;
    const int khalf = lane >> 4;          // 0 or 1
    const int mrow  = wave * 16 + m16;    // A-matrix row for this lane

    v8f acc_k[4]; v8f acc_v[4];
#pragma unroll
    for (int t = 0; t < 4; ++t) { acc_k[t] = (v8f)(0.f); acc_v[t] = (v8f)(0.f); }

#pragma unroll
    for (int c = 0; c < 2; ++c) {          // K chunks of 32
        // A fragment: elems 0..15 -> K = 32c + 16*khalf + e
        v16h a;
        const int kbA = c * 32 + khalf * 16;
#pragma unroll
        for (int e = 0; e < 8; ++e) {
            a[e]     = xn[mrow][kbA + e];
            a[e + 8] = xn[mrow][kbA + 8 + e];
        }
        const int kbB = c * 32 + khalf * 16;   // B: 16 contiguous K per lane half
#pragma unroll
        for (int t = 0; t < 4; ++t) {
            const int ncol = t * 16 + m16;
            v16h bk, bv;
#pragma unroll
            for (int e = 0; e < 16; ++e) {
                bk[e] = wkT[ncol][kbB + e];
                bv[e] = wvT[ncol][kbB + e];
            }
            acc_k[t] = __builtin_amdgcn_wmma_f32_16x16x32_f16(
                false, a, false, bk, (short)0, acc_k[t], false, false);
            acc_v[t] = __builtin_amdgcn_wmma_f32_16x16x32_f16(
                false, a, false, bv, (short)0, acc_v[t], false, false);
        }
    }

    __syncthreads();   // all waves done reading xn -> safe to reuse as k stage

    // ---- restage D tiles into LDS (lane l, vgpr j -> row j + 8*khalf) ----
#pragma unroll
    for (int t = 0; t < 4; ++t) {
#pragma unroll
        for (int j = 0; j < 8; ++j) {
            int mr = wave * 16 + j + khalf * 8;
            int nc = t * 16 + m16;
            xn[mr][nc]     = (_Float16)acc_k[t][j];
            vstage[mr][nc] = (_Float16)acc_v[t][j];
        }
    }
    __syncthreads();

    // ---- coalesced stores: 128 rows * 64 f16 = 16 KB each, 64 B per thread
    _Float16* kdst = k_out + (size_t)row0 * 64;
    _Float16* vdst = v_out + (size_t)row0 * 64;
#if HAVE_ASYNC_COPY
#pragma unroll
    for (int c2 = 0; c2 < 4; ++c2) {
        int off = (tid + 256 * c2) * 16;   // byte offset
        __builtin_amdgcn_global_store_async_from_lds_b128(
            (g_v4i*)((char*)kdst + off),
            (l_v4i*)((char*)&xn[0][0] + off), 0, 0);
        __builtin_amdgcn_global_store_async_from_lds_b128(
            (g_v4i*)((char*)vdst + off),
            (l_v4i*)((char*)&vstage[0][0] + off), 0, 0);
    }
    // ASYNCcnt drained by S_ENDPGM's implicit wait-idle
#else
    const uint4* ksrc = (const uint4*)&xn[0][0];
    const uint4* vsrc = (const uint4*)&vstage[0][0];
#pragma unroll
    for (int c2 = 0; c2 < 4; ++c2) {
        int idx = tid + 256 * c2;
        reinterpret_cast<uint4*>(kdst)[idx] = ksrc[idx];
        reinterpret_cast<uint4*>(vdst)[idx] = vsrc[idx];
    }
#endif
}

// ---------------------------------------------------------------------------
// Kernel 2 (per iter, 1 block): q = LN(slots)@Wq * D^-0.5 ; zero accumulators
// ---------------------------------------------------------------------------
__global__ __launch_bounds__(256) void q_kernel(
    const float* __restrict__ slots,
    const float* __restrict__ qfg_g, const float* __restrict__ qfg_b,
    const float* __restrict__ Wq_fg,
    const float* __restrict__ qbg_g, const float* __restrict__ qbg_b,
    const float* __restrict__ Wq_bg,
    float* __restrict__ q_out, float* __restrict__ colsum,
    float* __restrict__ upd_raw)
{
    __shared__ float lnv[4][64];
    __shared__ float red[256];
    const int tid = threadIdx.x;
    const int s = tid >> 6, d = tid & 63;

    float xv = slots[tid];
    red[tid] = xv; __syncthreads();
#pragma unroll
    for (int off = 32; off >= 1; off >>= 1) {
        if (d < off) red[tid] += red[tid + off];
        __syncthreads();
    }
    float mean = red[s * 64] * (1.f / 64.f);
    __syncthreads();
    float dv = xv - mean;
    red[tid] = dv * dv; __syncthreads();
#pragma unroll
    for (int off = 32; off >= 1; off >>= 1) {
        if (d < off) red[tid] += red[tid + off];
        __syncthreads();
    }
    float var = red[s * 64] * (1.f / 64.f);
    __syncthreads();

    const float* g = (s == 0) ? qbg_g : qfg_g;
    const float* b = (s == 0) ? qbg_b : qfg_b;
    lnv[s][d] = g[d] * dv * rsqrtf(var + LN_EPS) + b[d];
    __syncthreads();

    const float* Wq = (s == 0) ? Wq_bg : Wq_fg;
    float acc = 0.f;
#pragma unroll 8
    for (int j = 0; j < 64; ++j) acc += lnv[s][j] * Wq[j * 64 + d];
    q_out[tid] = acc * 0.125f;   // scale = 64^-0.5

    upd_raw[tid] = 0.f;
    if (tid < 4) colsum[tid] = 0.f;
}

// ---------------------------------------------------------------------------
// Kernel 3 (per iter): logits -> softmax -> attn.T out -> partial reductions
// 256 threads, 1 row per thread.  v rows staged to LDS with async copies
// overlapped with the logits/softmax computation.
// ---------------------------------------------------------------------------
__global__ __launch_bounds__(256) void attn_kernel(
    const _Float16* __restrict__ kmat, const _Float16* __restrict__ vmat,
    const float* __restrict__ q,
    float* __restrict__ attn_out,      // [4][N]
    float* __restrict__ colsum, float* __restrict__ upd_raw, int N)
{
    __shared__ float    qs[256];           // 4 x 64
    __shared__ float    wsh[4][256];       // per-row weights
    __shared__ _Float16 vsh[256][64];      // v rows (32 KB)

    const int tid = threadIdx.x;
    const size_t n = (size_t)blockIdx.x * 256 + tid;

    // kick off async v-row staging first so it overlaps the logits work
#if HAVE_ASYNC_COPY
    {
        const char* src = (const char*)(vmat + n * 64);
        char* dst = (char*)&vsh[tid][0];
#pragma unroll
        for (int i = 0; i < 8; ++i)
            __builtin_amdgcn_global_load_async_to_lds_b128(
                (g_v4i*)(src + i * 16), (l_v4i*)(dst + i * 16), 0, 0);
    }
#else
    {
        const uint4* src = reinterpret_cast<const uint4*>(vmat + n * 64);
        uint4* dst = reinterpret_cast<uint4*>(&vsh[tid][0]);
#pragma unroll
        for (int i = 0; i < 8; ++i) dst[i] = src[i];
    }
#endif

    qs[tid] = q[tid];
    __syncthreads();

    // load k row into registers (vectorized)
    _Float16 kr[64];
    {
        const uint4* src = reinterpret_cast<const uint4*>(kmat + n * 64);
        uint4* dst = reinterpret_cast<uint4*>(kr);
#pragma unroll
        for (int i = 0; i < 8; ++i) dst[i] = src[i];
    }
    float lg0 = 0.f, lg1 = 0.f, lg2 = 0.f, lg3 = 0.f;
#pragma unroll
    for (int j = 0; j < 64; ++j) {
        float kv = (float)kr[j];
        lg0 += kv * qs[j];
        lg1 += kv * qs[64 + j];
        lg2 += kv * qs[128 + j];
        lg3 += kv * qs[192 + j];
    }
    float mx = fmaxf(fmaxf(lg0, lg1), fmaxf(lg2, lg3));
    float e0 = __expf(lg0 - mx), e1 = __expf(lg1 - mx);
    float e2 = __expf(lg2 - mx), e3 = __expf(lg3 - mx);
    float inv = 1.f / (e0 + e1 + e2 + e3);
    float a0 = e0 * inv, a1 = e1 * inv, a2 = e2 * inv, a3 = e3 * inv;

    attn_out[0 * (size_t)N + n] = a0;
    attn_out[1 * (size_t)N + n] = a1;
    attn_out[2 * (size_t)N + n] = a2;
    attn_out[3 * (size_t)N + n] = a3;

    wsh[0][tid] = a0 + EPS_W; wsh[1][tid] = a1 + EPS_W;
    wsh[2][tid] = a2 + EPS_W; wsh[3][tid] = a3 + EPS_W;

#if HAVE_ASYNC_COPY
    wait_asynccnt0();          // v rows landed in LDS
#endif
    __syncthreads();

    // block reduction: thread (s,d) accumulates sum_i w[i][s] * v[i][d]
    const int s = tid >> 6, d = tid & 63;
    float acc = 0.f;
#pragma unroll 8
    for (int i = 0; i < 256; ++i) acc += wsh[s][i] * (float)vsh[i][d];
    atomicAdd(&upd_raw[s * 64 + d], acc);

    if (tid < 4) {
        float cs = 0.f;
#pragma unroll 8
        for (int i = 0; i < 256; ++i) cs += wsh[tid][i];
        atomicAdd(&colsum[tid], cs);
    }
}

// ---------------------------------------------------------------------------
// Kernel 4 (per iter, 1 block): normalize updates, GRU cell, LN + MLP residual
// ---------------------------------------------------------------------------
__device__ __forceinline__ float sigmoidf_(float x) { return 1.f / (1.f + __expf(-x)); }

__global__ __launch_bounds__(256) void slot_update_kernel(
    const float* __restrict__ colsum, const float* __restrict__ upd_raw,
    const float* __restrict__ fg_Wx, const float* __restrict__ fg_Wh,
    const float* __restrict__ fg_bin, const float* __restrict__ fg_brec,
    const float* __restrict__ bg_Wx, const float* __restrict__ bg_Wh,
    const float* __restrict__ bg_bin, const float* __restrict__ bg_brec,
    const float* __restrict__ mfg_g, const float* __restrict__ mfg_b,
    const float* __restrict__ mfg_W1, const float* __restrict__ mfg_b1,
    const float* __restrict__ mfg_W2, const float* __restrict__ mfg_b2,
    const float* __restrict__ mbg_g, const float* __restrict__ mbg_b,
    const float* __restrict__ mbg_W1, const float* __restrict__ mbg_b1,
    const float* __restrict__ mbg_W2, const float* __restrict__ mbg_b2,
    float* __restrict__ slots, float* __restrict__ slots_out)
{
    __shared__ float upd[4][64], hprev[4][64], lnv[4][64];
    __shared__ float gx[4][192], gh[4][192];
    __shared__ float hid[4][128];
    __shared__ float red[256];

    const int tid = threadIdx.x;
    const int s = tid >> 6, d = tid & 63;

    upd[s][d]   = upd_raw[tid] / colsum[s];
    hprev[s][d] = slots[tid];
    __syncthreads();

    // GRU gates: 768 dot products of length 64, 3 per thread
    for (int idx = tid; idx < 768; idx += 256) {
        int ss = idx / 192, gg = idx % 192;
        const float* Wx  = ss ? fg_Wx  : bg_Wx;
        const float* Wh  = ss ? fg_Wh  : bg_Wh;
        const float* bin = ss ? fg_bin : bg_bin;
        const float* brc = ss ? fg_brec: bg_brec;
        float ax = bin[gg], ah = brc[gg];
#pragma unroll 8
        for (int j = 0; j < 64; ++j) {
            ax += upd[ss][j]   * Wx[j * 192 + gg];
            ah += hprev[ss][j] * Wh[j * 192 + gg];
        }
        gx[ss][gg] = ax; gh[ss][gg] = ah;
    }
    __syncthreads();

    float z  = sigmoidf_(gx[s][d]       + gh[s][d]);
    float r  = sigmoidf_(gx[s][64 + d]  + gh[s][64 + d]);
    float hc = tanhf    (gx[s][128 + d] + r * gh[s][128 + d]);
    float hnew = z * hprev[s][d] + (1.f - z) * hc;

    // layernorm(hnew) per slot row
    red[tid] = hnew; __syncthreads();
#pragma unroll
    for (int off = 32; off >= 1; off >>= 1) {
        if (d < off) red[tid] += red[tid + off];
        __syncthreads();
    }
    float mean = red[s * 64] * (1.f / 64.f);
    __syncthreads();
    float dv = hnew - mean;
    red[tid] = dv * dv; __syncthreads();
#pragma unroll
    for (int off = 32; off >= 1; off >>= 1) {
        if (d < off) red[tid] += red[tid + off];
        __syncthreads();
    }
    float var = red[s * 64] * (1.f / 64.f);
    __syncthreads();

    const float* mg = s ? mfg_g : mbg_g;
    const float* mb = s ? mfg_b : mbg_b;
    lnv[s][d] = mg[d] * dv * rsqrtf(var + LN_EPS) + mb[d];
    __syncthreads();

    // MLP layer 1: 512 outputs of length-64 dots, relu
    for (int idx = tid; idx < 512; idx += 256) {
        int ss = idx >> 7, hh = idx & 127;
        const float* W1 = ss ? mfg_W1 : mbg_W1;
        const float* b1 = ss ? mfg_b1 : mbg_b1;
        float a = b1[hh];
#pragma unroll 8
        for (int j = 0; j < 64; ++j) a += lnv[ss][j] * W1[j * 128 + hh];
        hid[ss][hh] = fmaxf(a, 0.f);
    }
    __syncthreads();

    // MLP layer 2 + residual
    const float* W2 = s ? mfg_W2 : mbg_W2;
    const float* b2 = s ? mfg_b2 : mbg_b2;
    float o = b2[d];
#pragma unroll 8
    for (int j = 0; j < 128; ++j) o += hid[s][j] * W2[j * 64 + d];
    float out = hnew + o;
    slots[tid] = out;
    slots_out[tid] = out;
}

// ---------------------------------------------------------------------------
extern "C" void kernel_launch(void* const* d_in, const int* in_sizes, int n_in,
                              void* d_out, int out_size, void* d_ws, size_t ws_size,
                              hipStream_t stream) {
    const float* x        = (const float*)d_in[0];
    const float* noise_fg = (const float*)d_in[1];
    const float* noise_bg = (const float*)d_in[2];
    const float* ln_in_g  = (const float*)d_in[3];
    const float* ln_in_b  = (const float*)d_in[4];
    const float* mu_fg    = (const float*)d_in[5];
    const float* ls_fg    = (const float*)d_in[6];
    const float* mu_bg    = (const float*)d_in[7];
    const float* ls_bg    = (const float*)d_in[8];
    const float* Wk       = (const float*)d_in[9];
    const float* Wv       = (const float*)d_in[10];
    const float* qfg_g    = (const float*)d_in[11];
    const float* qfg_b    = (const float*)d_in[12];
    const float* Wq_fg    = (const float*)d_in[13];
    const float* qbg_g    = (const float*)d_in[14];
    const float* qbg_b    = (const float*)d_in[15];
    const float* Wq_bg    = (const float*)d_in[16];
    const float* fg_Wx    = (const float*)d_in[17];
    const float* fg_Wh    = (const float*)d_in[18];
    const float* fg_bin   = (const float*)d_in[19];
    const float* fg_brec  = (const float*)d_in[20];
    const float* bg_Wx    = (const float*)d_in[21];
    const float* bg_Wh    = (const float*)d_in[22];
    const float* bg_bin   = (const float*)d_in[23];
    const float* bg_brec  = (const float*)d_in[24];
    const float* mfg_g    = (const float*)d_in[25];
    const float* mfg_b    = (const float*)d_in[26];
    const float* mfg_W1   = (const float*)d_in[27];
    const float* mfg_b1   = (const float*)d_in[28];
    const float* mfg_W2   = (const float*)d_in[29];
    const float* mfg_b2   = (const float*)d_in[30];
    const float* mbg_g    = (const float*)d_in[31];
    const float* mbg_b    = (const float*)d_in[32];
    const float* mbg_W1   = (const float*)d_in[33];
    const float* mbg_b1   = (const float*)d_in[34];
    const float* mbg_W2   = (const float*)d_in[35];
    const float* mbg_b2   = (const float*)d_in[36];

    const int N = in_sizes[0] / 64;           // 262144

    // workspace layout
    char* w = (char*)d_ws;
    _Float16* k_h   = (_Float16*)w;                               // N*64 f16
    _Float16* v_h   = (_Float16*)(w + (size_t)N * 64 * 2);        // N*64 f16
    float*    slots = (float*)(w + (size_t)N * 64 * 4);           // 4*64
    float*    qbuf  = slots + 256;                                // 4*64
    float*    csum  = qbuf + 256;                                 // 4
    float*    uraw  = csum + 4;                                   // 4*64

    float* out_slots = (float*)d_out;
    float* out_attn  = out_slots + 256;

    init_slots_kernel<<<1, 256, 0, stream>>>(noise_fg, noise_bg, mu_fg, ls_fg,
                                             mu_bg, ls_bg, slots);
    kv_wmma_kernel<<<N / 128, 256, 0, stream>>>(x, ln_in_g, ln_in_b, Wk, Wv,
                                                k_h, v_h);
    for (int it = 0; it < 3; ++it) {
        q_kernel<<<1, 256, 0, stream>>>(slots, qfg_g, qfg_b, Wq_fg,
                                        qbg_g, qbg_b, Wq_bg, qbuf, csum, uraw);
        attn_kernel<<<N / 256, 256, 0, stream>>>(k_h, v_h, qbuf, out_attn,
                                                 csum, uraw, N);
        slot_update_kernel<<<1, 256, 0, stream>>>(
            csum, uraw,
            fg_Wx, fg_Wh, fg_bin, fg_brec,
            bg_Wx, bg_Wh, bg_bin, bg_brec,
            mfg_g, mfg_b, mfg_W1, mfg_b1, mfg_W2, mfg_b2,
            mbg_g, mbg_b, mbg_W1, mbg_b1, mbg_W2, mbg_b2,
            slots, out_slots);
    }
}
